// SparseAdaHGConv_25099788878230
// MI455X (gfx1250) — matrix operand order repaired and checked by reading, not compiled, not run on hardware
//
#include <hip/hip_runtime.h>
#include <math.h>

typedef unsigned short u16;
typedef unsigned int   u32;

typedef __attribute__((ext_vector_type(16))) __bf16 v16bf;
typedef __attribute__((ext_vector_type(8)))  float  v8f;

// round-to-nearest-even fp32 -> bf16
__device__ __forceinline__ u16 f2bf(float f) {
  u32 u = __float_as_uint(f);
  u32 r = (u + 0x7FFFu + ((u >> 16) & 1u)) >> 16;
  return (u16)r;
}

// ---------------------------------------------------------------- zero
__global__ void hg_zero_f4(float* __restrict__ p, int n4) {
  int i = blockIdx.x * blockDim.x + threadIdx.x;
  if (i < n4) {
    float4 z = {0.f, 0.f, 0.f, 0.f};
    *(float4*)(p + (size_t)i * 4) = z;
  }
}

// --------------------------------- W[k][n] fp32 -> Wt[n][k] bf16 (tiled)
__global__ __launch_bounds__(256) void hg_cvt_transpose(
    const float* __restrict__ W, u16* __restrict__ Wt) {
  __shared__ float t[32][33];
  const int n0 = blockIdx.x * 32;
  const int k0 = blockIdx.y * 32;
  const int c  = threadIdx.x & 31;
  const int r0 = threadIdx.x >> 5;        // 0..7
  for (int r = r0; r < 32; r += 8)
    t[r][c] = W[(size_t)(k0 + r) * 512 + n0 + c];
  __syncthreads();
  for (int r = r0; r < 32; r += 8)
    Wt[(size_t)(n0 + r) * 512 + k0 + c] = f2bf(t[c][r]);
}

// ---------------------------------------------------------------- scatter
// grid (D/64=8, N/512=8, B=4), block 256.  LDS partial He tile [256 edges][64 d].
__global__ __launch_bounds__(256) void hg_scatter(
    const float* __restrict__ X, const int* __restrict__ eidx,
    const float* __restrict__ ew, float* __restrict__ He) {
  __shared__ float part[256 * 64];   // 64 KB
  const int tid = threadIdx.x;
  for (int e = tid; e < 256 * 64; e += 256) part[e] = 0.f;
  __syncthreads();
  const int d0  = blockIdx.x * 64;
  const int n0  = blockIdx.y * 512;
  const int b   = blockIdx.z;
  const int dd  = tid & 63;
  const int sub = tid >> 6;          // 4 nodes in flight
  const size_t nb = (size_t)b * 4096;
  for (int i = 0; i < 512; i += 4) {
    const size_t row = nb + (n0 + i + sub);
    const float x = X[row * 512 + d0 + dd];
    const int*   ip = eidx + row * 8;
    const float* wp = ew   + row * 8;
#pragma unroll
    for (int j = 0; j < 8; ++j) {
      atomicAdd(&part[ip[j] * 64 + dd], wp[j] * x);   // ds_add_f32
    }
  }
  __syncthreads();
  float* Hb = He + ((size_t)b * 256) * 512 + d0;
  for (int e = tid; e < 256 * 64; e += 256) {
    atomicAdd(&Hb[(size_t)(e >> 6) * 512 + (e & 63)], part[e]);
  }
}

// ---------------------------------------------------------------- gather
__global__ __launch_bounds__(256) void hg_gather(
    const float* __restrict__ He, const int* __restrict__ eidx,
    const float* __restrict__ ew, float* __restrict__ Xmid) {
  const int tid = threadIdx.x;
  const int sub = tid >> 7;
  const int c4  = (tid & 127) * 4;
  const int ng  = blockIdx.x * 2 + sub;      // global node id
  const int b   = ng >> 12;
  const int*   ip = eidx + (size_t)ng * 8;
  const float* wp = ew   + (size_t)ng * 8;
  const float* Hb = He + (size_t)b * 256 * 512;
  float4 acc = {0.f, 0.f, 0.f, 0.f};
#pragma unroll
  for (int j = 0; j < 8; ++j) {
    const float wj = wp[j];
    const float4 h = *(const float4*)(Hb + (size_t)ip[j] * 512 + c4);
    acc.x += wj * h.x; acc.y += wj * h.y; acc.z += wj * h.z; acc.w += wj * h.w;
  }
  *(float4*)(Xmid + (size_t)ng * 512 + c4) = acc;
}

// ------------------------------------------- fused GEMM + GELU + LayerNorm
// Per workgroup: 32 rows x 512 cols.  8 waves; wave w owns cols [w*64,w*64+64)
// for BOTH 16-row blocks -> 8 WMMA tiles/wave/k-step, B fragments loaded
// straight from L2-resident pre-transposed bf16 W (two global_load_b128 each),
// A fragments from LDS.  No barriers inside the K loop.
struct __align__(16) HGSmem {
  union {
    u16   A[32 * 512];    // 32 KB : A rows (bf16) during GEMM
    float ln[32 * 512];   // 64 KB : post-GELU rows for LayerNorm
  };
};

union HGFrag { uint4 q[2]; v16bf v; };

__global__ __launch_bounds__(256) void hg_gemm_gelu_ln(
    const float* __restrict__ Xin, const u16* __restrict__ Wt,
    const float* __restrict__ bias, const float* __restrict__ gvec,
    const float* __restrict__ bvec, const float* __restrict__ resid,
    float* __restrict__ Xout) {
  __shared__ HGSmem sm;
  const int tid  = threadIdx.x;
  const int wv   = tid >> 5;
  const int lane = tid & 31;
  const int row0 = blockIdx.x * 32;

  // ---- stage A: 32x512 fp32 -> bf16 into LDS (float4 coalesced)
#pragma unroll
  for (int i = 0; i < 16; ++i) {
    const int e4 = (tid + 256 * i) * 4;
    const int r  = e4 >> 9;
    const int c  = e4 & 511;
    const float4 x = *(const float4*)(Xin + (size_t)(row0 + r) * 512 + c);
    ushort4 p;
    p.x = f2bf(x.x); p.y = f2bf(x.y); p.z = f2bf(x.z); p.w = f2bf(x.w);
    *(ushort4*)&sm.A[r * 512 + c] = p;
  }
  // warm W lines toward the caches while the barrier drains
  for (int i = tid; i < 512; i += 256)
    __builtin_prefetch(Wt + (size_t)i * 512, 0, 2);
  __syncthreads();

  v8f acc[2][4] = {};
  const int arow = lane & 15;
  const int hi   = (lane >> 4) << 3;   // A K-offset: 0 or 8
  const int bK   = (lane >> 4) << 3;   // B K-offset: 0 or 8
  const u16* wbase = Wt + (size_t)(wv * 64 + (lane & 15)) * 512;

  for (int kb = 0; kb < 16; ++kb) {
    const int k0 = kb * 32;
    // A fragments for both 16-row blocks (ISA 16x32 bf16 layout)
    HGFrag fa0, fa1;
    const u16* ap0 = &sm.A[arow * 512 + k0 + hi];
    fa0.q[0] = *(const uint4*)ap0;             // K: hi+0..7
    fa0.q[1] = *(const uint4*)(ap0 + 16);      // K: 16+hi+0..7
    const u16* ap1 = ap0 + 16 * 512;
    fa1.q[0] = *(const uint4*)ap1;
    fa1.q[1] = *(const uint4*)(ap1 + 16);
#pragma unroll
    for (int t = 0; t < 4; ++t) {
      HGFrag fb;                               // two 16-byte runs of one Wt row, 32 B apart
      const u16* bp = wbase + (size_t)t * 16 * 512 + k0 + bK;
      fb.q[0] = *(const uint4*)bp;             // K: bK+0..7
      fb.q[1] = *(const uint4*)(bp + 16);      // K: 16+bK+0..7
      acc[0][t] = __builtin_amdgcn_wmma_f32_16x16x32_bf16(
          false, fa0.v, false, fb.v, (short)0, acc[0][t], false, false);
      acc[1][t] = __builtin_amdgcn_wmma_f32_16x16x32_bf16(
          false, fa1.v, false, fb.v, (short)0, acc[1][t], false, false);
    }
  }
  __syncthreads();                             // A region free -> reuse as ln

  // ---- bias + GELU(erf), park rows in LDS for LayerNorm
  const int mofs = (lane >> 4) << 3;           // lanes>=16 hold M=8..15
#pragma unroll
  for (int rb = 0; rb < 2; ++rb) {
#pragma unroll
    for (int t = 0; t < 4; ++t) {
      const int col = wv * 64 + t * 16 + (lane & 15);
      const float bc = bias[col];
#pragma unroll
      for (int p = 0; p < 8; ++p) {
        const float v = acc[rb][t][p] + bc;
        const float g = 0.5f * v * (1.0f + erff(v * 0.70710678118654752440f));
        sm.ln[(rb * 16 + p + mofs) * 512 + col] = g;
      }
    }
  }
  __syncthreads();

  // ---- LayerNorm: each wave owns 4 rows
#pragma unroll
  for (int rr = 0; rr < 4; ++rr) {
    const int r = wv * 4 + rr;
    float s = 0.f, s2 = 0.f;
    for (int c = lane; c < 512; c += 32) {
      const float v = sm.ln[r * 512 + c];
      s += v; s2 += v * v;
    }
#pragma unroll
    for (int off = 16; off; off >>= 1) {
      s  += __shfl_xor(s,  off);
      s2 += __shfl_xor(s2, off);
    }
    const float mu   = s * (1.0f / 512.0f);
    const float var  = s2 * (1.0f / 512.0f) - mu * mu;
    const float rstd = rsqrtf(var + 1e-5f);
    const size_t gro = (size_t)(row0 + r) * 512;
    for (int c = lane; c < 512; c += 32) {
      float v = (sm.ln[r * 512 + c] - mu) * rstd * gvec[c] + bvec[c];
      if (resid) v += resid[gro + c];
      Xout[gro + c] = v;
    }
  }
}

// ================================================================ launch
extern "C" void kernel_launch(void* const* d_in, const int* in_sizes, int n_in,
                              void* d_out, int out_size, void* d_ws, size_t ws_size,
                              hipStream_t stream) {
  const float* X      = (const float*)d_in[0];
  const int*   eidx   = (const int*)  d_in[1];
  const float* ew     = (const float*)d_in[2];
  const float* W_edge = (const float*)d_in[3];
  const float* b_edge = (const float*)d_in[4];
  const float* g_edge = (const float*)d_in[5];
  const float* be_edge= (const float*)d_in[6];
  const float* W_node = (const float*)d_in[7];
  const float* b_node = (const float*)d_in[8];
  const float* g_node = (const float*)d_in[9];
  const float* be_node= (const float*)d_in[10];
  float* out = (float*)d_out;

  // workspace layout
  char* ws = (char*)d_ws;
  float* He   = (float*)ws;                                   // 2 MB
  float* Xmid = (float*)(ws + (size_t)2 * 1024 * 1024);       // 32 MB
  u16*   WtE  = (u16*)  (ws + (size_t)34 * 1024 * 1024);      // 512 KB
  u16*   WtN  = WtE + 512 * 512;                              // 512 KB

  // 1. zero He accumulator (4*256*512 floats)
  hg_zero_f4<<<512, 256, 0, stream>>>(He, 524288 / 4);
  // 2. transpose+convert weights to bf16 [n][k]
  dim3 tg(16, 16);
  hg_cvt_transpose<<<tg, 256, 0, stream>>>(W_edge, WtE);
  hg_cvt_transpose<<<tg, 256, 0, stream>>>(W_node, WtN);
  // 3. scatter nodes -> hyperedges
  dim3 sg(8, 8, 4);
  hg_scatter<<<sg, 256, 0, stream>>>(X, eidx, ew, He);
  // 4. edge projection: GEMM+GELU+LN, in-place on He (M = 1024 rows)
  hg_gemm_gelu_ln<<<1024 / 32, 256, 0, stream>>>(
      He, WtE, b_edge, g_edge, be_edge, nullptr, He);
  // 5. gather hyperedges -> nodes
  hg_gather<<<4 * 4096 / 2, 256, 0, stream>>>(He, eidx, ew, Xmid);
  // 6. node projection: GEMM+GELU+LN + residual X -> out (M = 16384 rows)
  hg_gemm_gelu_ln<<<16384 / 32, 256, 0, stream>>>(
      Xmid, WtN, b_node, g_node, be_node, X, out);
}
